// GPTLanguageModel_5806795784752
// MI455X (gfx1250) — matrix operand order repaired and checked by reading, not compiled, not run on hardware
//
#include <hip/hip_runtime.h>
#include <hip/hip_bf16.h>
#include <cstdint>

// ---------------------------------------------------------------------------
// Types for CDNA5 WMMA (wave32): v16bf A/B operands, v8f accumulator.
// ---------------------------------------------------------------------------
typedef __attribute__((ext_vector_type(16))) __bf16 bf16x16;
typedef __attribute__((ext_vector_type(8)))  float  floatx8;

union FragB16 { bf16x16 v; uint4 u[2]; };
static_assert(sizeof(bf16x16) == 32, "bf16x16 must be 32B");

// ---------------------------------------------------------------------------
// CDNA5 async global->LDS staging (ASYNCcnt-tracked), with portable fallback.
// The builtin expects: (AS1 int4*, AS3 int4*, imm offset, imm cpol).
// ---------------------------------------------------------------------------
#if defined(__gfx1250__) && __has_builtin(__builtin_amdgcn_global_load_async_to_lds_b128) && \
    __has_builtin(__builtin_amdgcn_s_wait_asynccnt)
#define USE_ASYNC_LDS 1
#else
#define USE_ASYNC_LDS 0
#endif

typedef int i32x4v __attribute__((vector_size(16)));
#if USE_ASYNC_LDS
typedef __attribute__((address_space(1))) i32x4v as1_i32x4;
typedef __attribute__((address_space(3))) i32x4v as3_i32x4;
#endif

__device__ __forceinline__ void stage16(const __bf16* g, __bf16* l) {
#if USE_ASYNC_LDS
  __builtin_amdgcn_global_load_async_to_lds_b128(
      (as1_i32x4*)g, (as3_i32x4*)l, 0, 0);
#else
  *(uint4*)l = *(const uint4*)g;
#endif
}
__device__ __forceinline__ void stage_wait() {
#if USE_ASYNC_LDS
  __builtin_amdgcn_s_wait_asynccnt(0);
#endif
}

// ---------------------------------------------------------------------------
// 16-lane xor butterflies: v_permlane16_b32 (VALU, no LDS round-trip) when
// available; falls back to __shfl_xor (ds_bpermute) otherwise.
// ---------------------------------------------------------------------------
#if defined(__gfx1250__) && __has_builtin(__builtin_amdgcn_permlane16)
#define HAVE_PERMLANE16 1
#else
#define HAVE_PERMLANE16 0
#endif

__host__ __device__ constexpr unsigned sel_half(int base, int M) {
  unsigned r = 0;
  for (int i = 0; i < 8; ++i)
    r |= (unsigned)(((base + i) ^ M) & 0xF) << (4 * i);
  return r;
}

template <int M>
__device__ __forceinline__ float lane_xor16(float x) {
#if HAVE_PERMLANE16
  unsigned u = __float_as_uint(x);
  unsigned r = __builtin_amdgcn_permlane16(u, u, sel_half(0, M), sel_half(8, M),
                                           false, false);
  return __uint_as_float(r);
#else
  return __shfl_xor(x, M, 32);
#endif
}

__device__ __forceinline__ float half_max(float v) {
  v = fmaxf(v, lane_xor16<1>(v));
  v = fmaxf(v, lane_xor16<2>(v));
  v = fmaxf(v, lane_xor16<4>(v));
  v = fmaxf(v, lane_xor16<8>(v));
  return v;
}
__device__ __forceinline__ float half_sum(float v) {
  v += lane_xor16<1>(v);
  v += lane_xor16<2>(v);
  v += lane_xor16<4>(v);
  v += lane_xor16<8>(v);
  return v;
}

// ---------------------------------------------------------------------------
// Embedding: x[b,t,:] = tok_emb[idx[b,t]] + pos_emb[t]
// ---------------------------------------------------------------------------
__global__ void embed_kernel(const int* __restrict__ idx,
                             const float* __restrict__ tok,
                             const float* __restrict__ pos,
                             float* __restrict__ x, int T, int D, size_t total) {
  size_t i = (size_t)blockIdx.x * blockDim.x + threadIdx.x;
  if (i >= total) return;
  size_t bt = i / D;
  int d = (int)(i % D);
  int t = (int)(bt % T);
  int token = idx[bt];
  x[i] = tok[(size_t)token * D + d] + pos[(size_t)t * D + d];
}

// ---------------------------------------------------------------------------
// LayerNorm fp32 -> bf16 (one 256-thread block per row, D=1024)
// ---------------------------------------------------------------------------
__global__ __launch_bounds__(256) void layernorm_kernel(
    const float* __restrict__ X, const float* __restrict__ g,
    const float* __restrict__ b, __bf16* __restrict__ out, int D) {
  __shared__ float s1[256], s2[256];
  int row = blockIdx.x, tid = threadIdx.x;
  const float* xr = X + (size_t)row * D;
  float sum = 0.f, sq = 0.f;
  for (int d = tid; d < D; d += 256) { float v = xr[d]; sum += v; sq += v * v; }
  s1[tid] = sum; s2[tid] = sq; __syncthreads();
  for (int o = 128; o > 0; o >>= 1) {
    if (tid < o) { s1[tid] += s1[tid + o]; s2[tid] += s2[tid + o]; }
    __syncthreads();
  }
  float mean = s1[0] / D;
  float var  = s2[0] / D - mean * mean;
  float r = rsqrtf(var + 1e-5f);
  __bf16* orow = out + (size_t)row * D;
  for (int d = tid; d < D; d += 256)
    orow[d] = (__bf16)((xr[d] - mean) * r * g[d] + b[d]);
}

// ---------------------------------------------------------------------------
// GEMM: C[M,N] = A(bf16)[M,K] * B(fp32 weights, converted inline)[K,N]
// 128x128 block tile, 8 waves, each wave 64x32 via 8x v_wmma_f32_16x16x32_bf16.
// Epilogue config is compile-time (no runtime pointer tests in the hot path).
// ---------------------------------------------------------------------------
#define TM 128
#define TN 128
#define TK 32

template <bool HAS_BIAS, bool HAS_RESID, bool RELU, bool OUT_F, bool OUT_H>
__global__ __launch_bounds__(256) void gemm_bf16f32(
    const __bf16* __restrict__ A, const float* __restrict__ Bw,
    const float* __restrict__ bias, const float* __restrict__ resid,
    float* __restrict__ outF, __bf16* __restrict__ outH,
    int M, int N, int K) {
  __shared__ __align__(16) __bf16 a_lds[TM][TK];   // 8 KB
  __shared__ __align__(16) __bf16 bt_lds[TN][TK];  // 8 KB (B transposed: [n][k])
  int tid  = threadIdx.x;
  int lane = tid & 31, wave = tid >> 5;
  int wm = (wave & 1) * 64;   // wave m-offset in tile
  int wn = (wave >> 1) * 32;  // wave n-offset in tile
  int m0 = blockIdx.y * TM, n0 = blockIdx.x * TN;
  int lm = lane & 15, hi = lane >> 4;
  int kbA = hi * 8;           // A operand K base (ISA 16-bit A layout)
  int kbB = hi * 8;           // B operand K base (same table order as A)

  floatx8 acc[4][2];
  #pragma unroll
  for (int i = 0; i < 4; ++i)
    #pragma unroll
    for (int j = 0; j < 2; ++j) acc[i][j] = (floatx8)0.0f;

  int ar = tid >> 1, aseg = (tid & 1) * 16;   // A: 2 threads/row, 16 bf16 each
  int bk = tid >> 3, bj = (tid & 7) * 16;     // B: 8 threads/k-row, 16 floats each
  const __bf16* arow = A + (size_t)(m0 + ar) * K + aseg;
  const float*  bsrc = Bw + (size_t)bk * N + n0 + bj;
  const size_t  bstep = (size_t)TK * N;

  for (int k0 = 0; k0 < K; k0 += TK) {
    // Stage A tile via CDNA5 async global->LDS copy (overlaps with B convert)
    stage16(arow,     &a_lds[ar][aseg]);
    stage16(arow + 8, &a_lds[ar][aseg + 8]);
    arow += TK;
    // Stage B tile: read fp32 weights once, convert to bf16, transpose to [n][k]
    if (k0 + TK < K)  // prefetch next weight k-tile into GL2 (global_prefetch_b8)
      __builtin_prefetch(bsrc + bstep, 0, 1);
    #pragma unroll
    for (int c = 0; c < 4; ++c) {
      float4 f = *(const float4*)(bsrc + c * 4);
      bt_lds[bj + c * 4 + 0][bk] = (__bf16)f.x;
      bt_lds[bj + c * 4 + 1][bk] = (__bf16)f.y;
      bt_lds[bj + c * 4 + 2][bk] = (__bf16)f.z;
      bt_lds[bj + c * 4 + 3][bk] = (__bf16)f.w;
    }
    bsrc += bstep;
    stage_wait();
    __syncthreads();

    FragB16 af[4], bfr[2];
    #pragma unroll
    for (int i = 0; i < 4; ++i) {
      const __bf16* p = &a_lds[wm + i * 16 + lm][0];
      af[i].u[0] = *(const uint4*)(p + kbA);
      af[i].u[1] = *(const uint4*)(p + kbA + 16);
    }
    #pragma unroll
    for (int j = 0; j < 2; ++j) {
      const __bf16* p = &bt_lds[wn + j * 16 + lm][kbB];
      bfr[j].u[0] = *(const uint4*)(p);
      bfr[j].u[1] = *(const uint4*)(p + 16);
    }
    #pragma unroll
    for (int i = 0; i < 4; ++i)
      #pragma unroll
      for (int j = 0; j < 2; ++j)
        acc[i][j] = __builtin_amdgcn_wmma_f32_16x16x32_bf16(
            false, af[i].v, false, bfr[j].v, (short)0, acc[i][j], false, false);
    __syncthreads();
  }

  // Epilogue (C/D layout: lane -> N = lm, VGPR v -> M = v + 8*hi)
  #pragma unroll
  for (int j = 0; j < 2; ++j) {
    int n = n0 + wn + j * 16 + lm;
    float bv = HAS_BIAS ? bias[n] : 0.0f;
    #pragma unroll
    for (int i = 0; i < 4; ++i) {
      #pragma unroll
      for (int v = 0; v < 8; ++v) {
        int m = m0 + wm + i * 16 + v + 8 * hi;
        size_t off = (size_t)m * N + n;
        float val = acc[i][j][v] + bv;
        if constexpr (HAS_RESID) val += resid[off];
        if constexpr (RELU)      val = fmaxf(val, 0.0f);
        if constexpr (OUT_F)     outF[off] = val;
        if constexpr (OUT_H)     outH[off] = (__bf16)val;
      }
    }
  }
}

// ---------------------------------------------------------------------------
// Flash-style causal attention, HD=64. One block per (b, h, 128 queries);
// 8 waves each own a 16-query tile; K/V 32-key tiles async-staged in LDS,
// shared by all waves (uniform trip count -> convergent barriers).
// Online softmax kept in registers; P transposed through per-wave LDS scratch.
// ---------------------------------------------------------------------------
#define AHD 64
__global__ __launch_bounds__(256) void attn_kernel(
    const __bf16* __restrict__ Q, const __bf16* __restrict__ Kb,
    const __bf16* __restrict__ Vb, __bf16* __restrict__ O,
    int T, int Dm, int Hh, float scale) {
  __shared__ __align__(16) __bf16 kt_lds[32][AHD];     // 4 KB
  __shared__ __align__(16) __bf16 vt_lds[32][AHD];     // 4 KB
  __shared__ __align__(16) __bf16 p_lds[8][16][32];    // 8 KB (per-wave P scratch)
  int tid = threadIdx.x, lane = tid & 31, wave = tid >> 5;
  int lm = lane & 15, hi = lane >> 4;
  int kbA = hi * 8, kbB = hi * 8;
  int nqb = T / 128;
  int qblk = blockIdx.x % nqb;
  int bh   = blockIdx.x / nqb;
  int hh = bh % Hh, bb = bh / Hh;
  size_t brow = (size_t)bb * T;
  int qbase = qblk * 128 + wave * 16;

  FragB16 qa[2];  // Q[16x64] = two A-operands (K-chunks of 32)
  {
    const __bf16* qrow = Q + (brow + qbase + lm) * Dm + hh * AHD;
    #pragma unroll
    for (int c = 0; c < 2; ++c) {
      qa[c].u[0] = *(const uint4*)(qrow + c * 32 + kbA);
      qa[c].u[1] = *(const uint4*)(qrow + c * 32 + kbA + 16);
    }
  }
  floatx8 oacc[4];
  #pragma unroll
  for (int t = 0; t < 4; ++t) oacc[t] = (floatx8)0.0f;
  float rm[8], rs[8];
  #pragma unroll
  for (int v = 0; v < 8; ++v) { rm[v] = -1e30f; rs[v] = 0.f; }

  int cr = tid >> 3, cseg = (tid & 7) * 8;  // cooperative K/V staging (16B each)
  const __bf16* kptr = Kb + (brow + cr) * Dm + hh * AHD + cseg;
  const __bf16* vptr = Vb + (brow + cr) * Dm + hh * AHD + cseg;
  const size_t  kvstep = (size_t)32 * Dm;

  int nkt = qblk * 4 + 4;                   // key tiles needed for this block
  for (int kt = 0; kt < nkt; ++kt) {
    int kbase = kt * 32;
    stage16(kptr, &kt_lds[cr][cseg]);
    stage16(vptr, &vt_lds[cr][cseg]);
    kptr += kvstep; vptr += kvstep;
    stage_wait();
    __syncthreads();

    // S[16q x 32k] = Q * K^T : two 16x16 score tiles, each chaining HD=64 as 2 WMMAs
    floatx8 s[2];
    #pragma unroll
    for (int t2 = 0; t2 < 2; ++t2) {
      s[t2] = (floatx8)0.0f;
      #pragma unroll
      for (int c = 0; c < 2; ++c) {
        FragB16 kf;  // B operand: lane column = key (t2*16+lm), K = HD dims
        const __bf16* p = &kt_lds[t2 * 16 + lm][c * 32 + kbB];
        kf.u[0] = *(const uint4*)(p);
        kf.u[1] = *(const uint4*)(p + 16);
        s[t2] = __builtin_amdgcn_wmma_f32_16x16x32_bf16(
            false, qa[c].v, false, kf.v, (short)0, s[t2], false, false);
      }
    }

    // Online softmax update (row reductions across 16-lane halves, permlane16)
    #pragma unroll
    for (int v = 0; v < 8; ++v) {
      int qi = qbase + v + 8 * hi;
      float s1 = s[0][v] * scale; if (kbase + lm > qi)      s1 = -1e30f;
      float s2 = s[1][v] * scale; if (kbase + 16 + lm > qi) s2 = -1e30f;
      float tmax = half_max(fmaxf(s1, s2));
      float nm = fmaxf(rm[v], tmax);
      float alpha = __expf(rm[v] - nm);
      float p1 = __expf(s1 - nm), p2 = __expf(s2 - nm);
      float ts = half_sum(p1 + p2);
      rs[v] = rs[v] * alpha + ts;
      rm[v] = nm;
      #pragma unroll
      for (int nt = 0; nt < 4; ++nt) oacc[nt][v] *= alpha;
      int prow = v + 8 * hi;
      p_lds[wave][prow][lm]      = (__bf16)p1;
      p_lds[wave][prow][lm + 16] = (__bf16)p2;
    }
    __syncthreads();

    // O[16x64] += P[16x32] * V[32x64]
    FragB16 pf;
    pf.u[0] = *(const uint4*)&p_lds[wave][lm][kbA];
    pf.u[1] = *(const uint4*)&p_lds[wave][lm][kbA + 16];
    #pragma unroll
    for (int nt = 0; nt < 4; ++nt) {
      FragB16 vf;  // B operand: lane column = dim nt*16+lm, K = keys
      #pragma unroll
      for (int j = 0; j < 16; ++j) vf.v[j] = vt_lds[kbB + j + (j & 8)][nt * 16 + lm];
      oacc[nt] = __builtin_amdgcn_wmma_f32_16x16x32_bf16(
          false, pf.v, false, vf.v, (short)0, oacc[nt], false, false);
    }
    __syncthreads();
  }

  #pragma unroll
  for (int v = 0; v < 8; ++v) {
    float inv = 1.0f / rs[v];
    int m = qbase + v + 8 * hi;
    #pragma unroll
    for (int nt = 0; nt < 4; ++nt)
      O[(brow + m) * Dm + hh * AHD + nt * 16 + lm] = (__bf16)(oacc[nt][v] * inv);
  }
}

// ---------------------------------------------------------------------------
// Per-row NLL over V=32000 (one block per row), then deterministic mean.
// ---------------------------------------------------------------------------
__global__ __launch_bounds__(256) void nll_kernel(
    const float* __restrict__ logits, const int* __restrict__ tgt,
    float* __restrict__ nll, int Vv) {
  __shared__ float sh[256];
  int row = blockIdx.x, tid = threadIdx.x;
  const float* lr = logits + (size_t)row * Vv;
  float mx = -3.4e38f;
  for (int j = tid; j < Vv; j += 256) mx = fmaxf(mx, lr[j]);
  sh[tid] = mx; __syncthreads();
  for (int o = 128; o > 0; o >>= 1) {
    if (tid < o) sh[tid] = fmaxf(sh[tid], sh[tid + o]);
    __syncthreads();
  }
  mx = sh[0]; __syncthreads();
  float sum = 0.f;
  for (int j = tid; j < Vv; j += 256) sum += __expf(lr[j] - mx);
  sh[tid] = sum; __syncthreads();
  for (int o = 128; o > 0; o >>= 1) {
    if (tid < o) sh[tid] += sh[tid + o];
    __syncthreads();
  }
  if (tid == 0) nll[row] = (mx + __logf(sh[0])) - lr[tgt[row]];
}

__global__ __launch_bounds__(256) void loss_kernel(
    const float* __restrict__ nll, int n, float* __restrict__ out) {
  __shared__ float sh[256];
  int tid = threadIdx.x;
  float s = 0.f;
  for (int i = tid; i < n; i += 256) s += nll[i];
  sh[tid] = s; __syncthreads();
  for (int o = 128; o > 0; o >>= 1) {
    if (tid < o) sh[tid] += sh[tid + o];
    __syncthreads();
  }
  if (tid == 0) *out = sh[0] / n;
}

// ---------------------------------------------------------------------------
// Host orchestration
// ---------------------------------------------------------------------------
extern "C" void kernel_launch(void* const* d_in, const int* in_sizes, int n_in,
                              void* d_out, int out_size, void* d_ws, size_t ws_size,
                              hipStream_t stream) {
  (void)in_sizes; (void)n_in; (void)out_size; (void)ws_size;
  constexpr int Lc = 6, Hc = 16, Dc = 1024, Vc = 32000, Tc = 2048, Bc = 2;
  constexpr int Mc = Bc * Tc;   // 4096 token rows
  constexpr int Fc = 4 * Dc;    // 4096

  const int*   idx     = (const int*)  d_in[0];
  const int*   targets = (const int*)  d_in[1];
  const float* tok_emb = (const float*)d_in[2];
  const float* pos_emb = (const float*)d_in[3];
  const float* wq      = (const float*)d_in[4];
  const float* wk      = (const float*)d_in[5];
  const float* wv      = (const float*)d_in[6];
  const float* w_proj  = (const float*)d_in[7];
  const float* b_proj  = (const float*)d_in[8];
  const float* w_fc1   = (const float*)d_in[9];
  const float* b_fc1   = (const float*)d_in[10];
  const float* w_fc2   = (const float*)d_in[11];
  const float* b_fc2   = (const float*)d_in[12];
  const float* ln1_g   = (const float*)d_in[13];
  const float* ln1_b   = (const float*)d_in[14];
  const float* ln2_g   = (const float*)d_in[15];
  const float* ln2_b   = (const float*)d_in[16];
  const float* lnf_g   = (const float*)d_in[17];
  const float* lnf_b   = (const float*)d_in[18];
  const float* w_head  = (const float*)d_in[19];
  const float* b_head  = (const float*)d_in[20];

  // Workspace layout (~92 MB total)
  char* wp = (char*)d_ws;
  auto take = [&](size_t bytes) -> char* {
    char* p = wp; wp += (bytes + 255) & ~(size_t)255; return p;
  };
  float*  x   = (float*) take((size_t)Mc * Dc * sizeof(float)); // residual stream fp32
  __bf16* h   = (__bf16*)take((size_t)Mc * Dc * 2);             // LN output
  __bf16* qb  = (__bf16*)take((size_t)Mc * Dc * 2);
  __bf16* kb  = (__bf16*)take((size_t)Mc * Dc * 2);
  __bf16* vb  = (__bf16*)take((size_t)Mc * Dc * 2);
  __bf16* ob  = (__bf16*)take((size_t)Mc * Dc * 2);             // attention out
  __bf16* fb  = (__bf16*)take((size_t)Mc * Fc * 2);             // MLP hidden
  float*  nll = (float*) take((size_t)Mc * sizeof(float));

  float* logits = (float*)d_out;                 // [Mc, Vc] fp32
  float* loss   = logits + (size_t)Mc * Vc;      // scalar

  dim3 blk(256);
  size_t total = (size_t)Mc * Dc;
  embed_kernel<<<(unsigned)((total + 255) / 256), blk, 0, stream>>>(
      idx, tok_emb, pos_emb, x, Tc, Dc, total);

  const float scale = 0.125f;  // HD^-0.5, HD=64
  dim3 gDD(Dc / 128, Mc / 128);
  dim3 gF1(Fc / 128, Mc / 128);
  for (int l = 0; l < Lc; ++l) {
    layernorm_kernel<<<Mc, blk, 0, stream>>>(
        x, ln1_g + (size_t)l * Dc, ln1_b + (size_t)l * Dc, h, Dc);
    // q/k/v: plain GEMM -> bf16
    gemm_bf16f32<false, false, false, false, true><<<gDD, blk, 0, stream>>>(
        h, wq + (size_t)l * Dc * Dc, nullptr, nullptr, nullptr, qb, Mc, Dc, Dc);
    gemm_bf16f32<false, false, false, false, true><<<gDD, blk, 0, stream>>>(
        h, wk + (size_t)l * Dc * Dc, nullptr, nullptr, nullptr, kb, Mc, Dc, Dc);
    gemm_bf16f32<false, false, false, false, true><<<gDD, blk, 0, stream>>>(
        h, wv + (size_t)l * Dc * Dc, nullptr, nullptr, nullptr, vb, Mc, Dc, Dc);
    attn_kernel<<<Bc * Hc * (Tc / 128), blk, 0, stream>>>(
        qb, kb, vb, ob, Tc, Dc, Hc, scale);
    // proj: bias + residual -> fp32 x
    gemm_bf16f32<true, true, false, true, false><<<gDD, blk, 0, stream>>>(
        ob, w_proj + (size_t)l * Dc * Dc, b_proj + (size_t)l * Dc, x, x, nullptr,
        Mc, Dc, Dc);
    layernorm_kernel<<<Mc, blk, 0, stream>>>(
        x, ln2_g + (size_t)l * Dc, ln2_b + (size_t)l * Dc, h, Dc);
    // fc1: bias + relu -> bf16
    gemm_bf16f32<true, false, true, false, true><<<gF1, blk, 0, stream>>>(
        h, w_fc1 + (size_t)l * Dc * Fc, b_fc1 + (size_t)l * Fc, nullptr, nullptr, fb,
        Mc, Fc, Dc);
    // fc2: bias + residual -> fp32 x
    gemm_bf16f32<true, true, false, true, false><<<gDD, blk, 0, stream>>>(
        fb, w_fc2 + (size_t)l * Fc * Dc, b_fc2 + (size_t)l * Dc, x, x, nullptr,
        Mc, Dc, Fc);
  }
  layernorm_kernel<<<Mc, blk, 0, stream>>>(x, lnf_g, lnf_b, h, Dc);
  // head: bias -> fp32 logits (straight into d_out)
  dim3 gHead(Vc / 128, Mc / 128);
  gemm_bf16f32<true, false, false, true, false><<<gHead, blk, 0, stream>>>(
      h, w_head, b_head, nullptr, logits, nullptr, Mc, Vc, Dc);
  nll_kernel<<<Mc, blk, 0, stream>>>(logits, targets, nll, Vc);
  loss_kernel<<<1, blk, 0, stream>>>(nll, Mc, loss);
}
